// Network_86139864089438
// MI455X (gfx1250) — compile-verified, matrix operand order checked by reading, not run on hardware
//
#include <hip/hip_runtime.h>
#include <cstdint>
#include <cstddef>

// ---------------------------------------------------------------------------
// Mamba (4 layers) on MI455X / gfx1250.
//
// GEMMs: v_wmma_f32_16x16x32_bf16, fp32 accumulate. All GEMM operands are
// pre-converted to bf16 exactly once (weights in a one-time pre-pass, padded
// to N%64 / K%32 with zeros so unpredicated DMA tiles stay in-bounds and
// mathematically inert; activations emitted as bf16 by their producer
// kernels). GEMM tiles are staged straight into LDS with the CDNA5 async DMA
// path (global_load_async_to_lds_b128 / s_wait_asynccnt), double-buffered so
// the next K-tile streams while the matrix pipe consumes the current one.
//
// Working set (~105MB) is L2-resident (192MB): GEMMs are matrix-pipe bound,
// the selective scan is the sequential part (4096 independent (b,d) channels
// = 128 wave32 waves hide its latency).
//
// NOTE: the reference's out_proj weights (_OUTW) are module-level Python
// constants, absent from the input list; deterministic placeholders are
// synthesized in workspace (compile-only environment).
// ---------------------------------------------------------------------------

#define DM      256
#define DI      512
#define DSTATE  16
#define NB      8
#define LL      1024
#define ROWS    (NB * LL)        // 8192 tokens

typedef __bf16 bf16_t;
typedef bf16_t v16bf __attribute__((ext_vector_type(16)));
typedef float  v8f   __attribute__((ext_vector_type(8)));

union FragU { v16bf v; uint4 q[2]; };

__device__ __forceinline__ float sigmoidf_(float x) {
    return 1.0f / (1.0f + __expf(-x));
}

// ---- CDNA5 async DMA helpers (inline asm: portable across toolchains) -----
__device__ __forceinline__ void async_b128(unsigned lds_byte, const bf16_t* g) {
    // per-lane: LDS[lds_byte] = MEM[g .. g+15]; tracked by ASYNCcnt
    asm volatile("global_load_async_to_lds_b128 %0, %1, off"
                 :: "v"(lds_byte), "v"(g) : "memory");
}
__device__ __forceinline__ void wait_async0() {
    asm volatile("s_wait_asynccnt 0x0" ::: "memory");
}
__device__ __forceinline__ void wait_async3() {
    asm volatile("s_wait_asynccnt 0x3" ::: "memory");
}

// ---------------------------------------------------------------------------
// WMMA GEMM:  C[m,n] = act( sum_k A[m,k]*B[n,k] + bias[n] ) + add[m,n]
//   A: (M, lda) bf16 row-major; B: (N, ldb) bf16 row-major (zero-padded).
//   K must be a multiple of 32 (caller passes padded K; pad cols of B are 0).
// Block: 256 thr = 8 waves; block tile 128(M) x 64(N); wave tile 32x32
// (2x2 wmma accumulators). Double-buffered async-DMA staging: per stage each
// wave issues 2 async b128 ops for the A tile + 1 for the B tile.
// act: 0 = none, 1 = softplus. Cbf: optional bf16 mirror of C.
// ---------------------------------------------------------------------------
__global__ __launch_bounds__(256) void gemm_bf16_wmma(
    const bf16_t* __restrict__ A, int lda,
    const bf16_t* __restrict__ Bw, int ldb,
    const float* __restrict__ bias,
    const float* __restrict__ add,
    float* __restrict__ C, int ldc,
    bf16_t* __restrict__ Cbf,
    int M, int N, int K, int act)
{
    // [buf][row][20 uints] : 32 bf16 (16 uints) + 4 pad uints (80B row pitch)
    __shared__ __align__(16) unsigned As[2][128][20];
    __shared__ __align__(16) unsigned Bs[2][64][20];

    const int t    = threadIdx.x;
    const int bm   = blockIdx.y * 128;
    const int bn   = blockIdx.x * 64;
    const int lane = t & 31;
    const int wid  = t >> 5;
    const int wm   = (wid & 3) * 32;
    const int wn   = (wid >> 2) * 32;

    const unsigned asBase = (unsigned)(uintptr_t)(&As[0][0][0]);
    const unsigned bsBase = (unsigned)(uintptr_t)(&Bs[0][0][0]);

    v8f acc[2][2];
    acc[0][0] = {}; acc[0][1] = {}; acc[1][0] = {}; acc[1][1] = {};

    const bf16_t* Abase = A + (size_t)bm * lda;
    const bf16_t* Bbase = Bw + (size_t)bn * ldb;

    // per-thread DMA chunk assignments (16B chunks)
    const int ar0 = t >> 2,          af0 = t & 3;          // A chunk t
    const int ar1 = (t + 256) >> 2,  af1 = (t + 256) & 3;  // A chunk t+256
    const int brr = t >> 2,          bff = t & 3;          // B chunk t

    auto stage = [&](int buf, int kt) {
        const bf16_t* Ak = Abase + kt;
        const bf16_t* Bk = Bbase + kt;
        unsigned ab = asBase + (unsigned)buf * 10240u;
        unsigned bb = bsBase + (unsigned)buf * 5120u;
        async_b128(ab + (unsigned)ar0 * 80u + (unsigned)af0 * 16u,
                   Ak + (size_t)ar0 * lda + af0 * 8);
        async_b128(ab + (unsigned)ar1 * 80u + (unsigned)af1 * 16u,
                   Ak + (size_t)ar1 * lda + af1 * 8);
        async_b128(bb + (unsigned)brr * 80u + (unsigned)bff * 16u,
                   Bk + (size_t)brr * ldb + bff * 8);
    };

    const int nT = K >> 5;
    stage(0, 0);
    for (int i = 0; i < nT; ++i) {
        const int cur = i & 1;
        if (i + 1 < nT) {
            stage(cur ^ 1, (i + 1) << 5);
            wait_async3();                 // stage i complete (3 newer in flight)
        } else {
            wait_async0();                 // final stage complete
        }
        __syncthreads();                   // publish LDS tile to all waves

        const int mrow = lane & 15;
        const int khu  = (lane >> 4) << 2; // uint col offset: 0 or 4
        FragU a0, a1, b0, b1;
        {
            const unsigned* r0 = As[cur][wm + mrow];
            a0.q[0] = *(const uint4*)(r0 + khu);
            a0.q[1] = *(const uint4*)(r0 + khu + 8);
            const unsigned* r1 = As[cur][wm + 16 + mrow];
            a1.q[0] = *(const uint4*)(r1 + khu);
            a1.q[1] = *(const uint4*)(r1 + khu + 8);
            const unsigned* c0 = Bs[cur][wn + mrow];
            b0.q[0] = *(const uint4*)(c0 + khu);
            b0.q[1] = *(const uint4*)(c0 + khu + 8);
            const unsigned* c1 = Bs[cur][wn + 16 + mrow];
            b1.q[0] = *(const uint4*)(c1 + khu);
            b1.q[1] = *(const uint4*)(c1 + khu + 8);
        }
        acc[0][0] = __builtin_amdgcn_wmma_f32_16x16x32_bf16(false, a0.v, false, b0.v, (short)0, acc[0][0], false, false);
        acc[0][1] = __builtin_amdgcn_wmma_f32_16x16x32_bf16(false, a0.v, false, b1.v, (short)0, acc[0][1], false, false);
        acc[1][0] = __builtin_amdgcn_wmma_f32_16x16x32_bf16(false, a1.v, false, b0.v, (short)0, acc[1][0], false, false);
        acc[1][1] = __builtin_amdgcn_wmma_f32_16x16x32_bf16(false, a1.v, false, b1.v, (short)0, acc[1][1], false, false);
        __syncthreads();                   // tile consumed; safe to overwrite
    }

    // epilogue: VGPR r, lanes 0-15 -> (M=r, N=lane); lanes 16-31 -> (M=r+8)
    const int nlane = lane & 15;
    const int mhalf = (lane >> 4) * 8;
    #pragma unroll
    for (int mt = 0; mt < 2; ++mt) {
        #pragma unroll
        for (int nt = 0; nt < 2; ++nt) {
            int n = bn + wn + nt * 16 + nlane;
            if (n >= N) continue;
            float bv = bias ? bias[n] : 0.0f;
            #pragma unroll
            for (int r = 0; r < 8; ++r) {
                int m = bm + wm + mt * 16 + mhalf + r;
                float v = acc[mt][nt][r] + bv;
                if (act == 1) v = (v > 20.0f) ? v : log1pf(__expf(v));  // softplus
                if (add) v += add[(size_t)m * ldc + n];
                C[(size_t)m * ldc + n] = v;
                if (Cbf) Cbf[(size_t)m * ldc + n] = (bf16_t)v;
            }
        }
    }
}

// ---------------------------------------------------------------------------
// Weight pre-conversion fp32 (N,K) -> bf16 (Npad,Kpad), zero-padded.
// ---------------------------------------------------------------------------
__global__ void cvt_weight_kernel(const float* __restrict__ src, bf16_t* __restrict__ dst,
                                  int N, int K, int Npad, int Kpad)
{
    int idx = blockIdx.x * blockDim.x + threadIdx.x;
    if (idx >= Npad * Kpad) return;
    int n = idx / Kpad, k = idx - n * Kpad;
    dst[idx] = (n < N && k < K) ? (bf16_t)src[(size_t)n * K + k] : (bf16_t)0.0f;
}

// ---------------------------------------------------------------------------
// RMSNorm over rows of 256 -> bf16 output (GEMM A operand).
// ---------------------------------------------------------------------------
__global__ __launch_bounds__(256) void rmsnorm_kernel(
    const float* __restrict__ x, const float* __restrict__ w, bf16_t* __restrict__ o)
{
    __shared__ float red[8];
    __shared__ float scl;
    int row = blockIdx.x, t = threadIdx.x;
    float v  = x[(size_t)row * DM + t];
    float ss = v * v;
    for (int d = 16; d; d >>= 1) ss += __shfl_xor(ss, d, 32);
    if ((t & 31) == 0) red[t >> 5] = ss;
    __syncthreads();
    if (t == 0) {
        float s = 0.0f;
        for (int i = 0; i < 8; ++i) s += red[i];
        scl = rsqrtf(s / (float)DM + 1e-5f);
    }
    __syncthreads();
    o[(size_t)row * DM + t] = (bf16_t)(v * scl * w[t]);
}

// ---------------------------------------------------------------------------
// Depthwise causal conv1d (width 4) + bias + SiLU. Emits fp32 (for the scan)
// and bf16 (x_proj GEMM A operand).
// ---------------------------------------------------------------------------
__global__ __launch_bounds__(256) void conv_silu_kernel(
    const float* __restrict__ xz, const float* __restrict__ cw,
    const float* __restrict__ cb, float* __restrict__ xs, bf16_t* __restrict__ xsb)
{
    int idx = blockIdx.x * blockDim.x + threadIdx.x;   // (b*LL + l)*DI + d
    int d = idx & (DI - 1);
    int l = (idx >> 9) & (LL - 1);
    int b = idx >> 19;
    float acc = cb[d];
    const float* col = xz + (size_t)b * LL * (2 * DI) + d;
    #pragma unroll
    for (int j = 0; j < 4; ++j) {
        int ls = l - 3 + j;
        if (ls >= 0) acc += cw[d * 4 + j] * col[(size_t)ls * (2 * DI)];
    }
    float v = acc * sigmoidf_(acc);                    // SiLU
    xs[(size_t)idx]  = v;
    xsb[(size_t)idx] = (bf16_t)v;
}

// ---------------------------------------------------------------------------
// Selective scan fused with output gate y * silu(res); bf16 output (GEMM A).
// One lane per (b,d) channel; 16-state recurrence in registers; per-step B/C
// vectors staged in LDS.
// ---------------------------------------------------------------------------
__global__ __launch_bounds__(256) void scan_kernel(
    const float* __restrict__ xs, const float* __restrict__ dlt,
    const float* __restrict__ xdbl, const float* __restrict__ xz,
    const float* __restrict__ A_log, const float* __restrict__ Dp,
    bf16_t* __restrict__ y)
{
    __shared__ float sBC[32];                  // [0..15]=B, [16..31]=C
    int b = blockIdx.x >> 1;
    int d = ((blockIdx.x & 1) << 8) + threadIdx.x;

    float Ad[DSTATE];
    #pragma unroll
    for (int n = 0; n < DSTATE; ++n) Ad[n] = -__expf(A_log[d * DSTATE + n]);
    float hst[DSTATE];
    #pragma unroll
    for (int n = 0; n < DSTATE; ++n) hst[n] = 0.0f;
    float Dpd = Dp[d];

    for (int l = 0; l < LL; ++l) {
        size_t row = (size_t)b * LL + l;
        if (threadIdx.x < 32) sBC[threadIdx.x] = xdbl[row * 48 + 16 + threadIdx.x];
        __syncthreads();
        float u  = xs[row * DI + d];
        float dl = dlt[row * DI + d];
        float yv = 0.0f;
        #pragma unroll
        for (int n = 0; n < DSTATE; ++n) {
            float dA = __expf(dl * Ad[n]);
            hst[n] = dA * hst[n] + (dl * sBC[n]) * u;
            yv += hst[n] * sBC[16 + n];
        }
        float res = xz[row * (2 * DI) + DI + d];
        y[row * DI + d] = (bf16_t)((yv + u * Dpd) * (res * sigmoidf_(res)));
        __syncthreads();
    }
}

// ---------------------------------------------------------------------------
// Final: rmsnorm + lin_out (D_OUT = 1) + LeakyReLU(0.01). One block per row.
// ---------------------------------------------------------------------------
__global__ __launch_bounds__(256) void final_kernel(
    const float* __restrict__ h, const float* __restrict__ nw,
    const float* __restrict__ w0, const float* __restrict__ b0,
    float* __restrict__ out)
{
    __shared__ float red[8];
    __shared__ float scl;
    int row = blockIdx.x, t = threadIdx.x;
    float v  = h[(size_t)row * DM + t];
    float ss = v * v;
    for (int d = 16; d; d >>= 1) ss += __shfl_xor(ss, d, 32);
    if ((t & 31) == 0) red[t >> 5] = ss;
    __syncthreads();
    if (t == 0) {
        float s = 0.0f;
        for (int i = 0; i < 8; ++i) s += red[i];
        scl = rsqrtf(s / (float)DM + 1e-5f);
    }
    __syncthreads();
    float c = v * scl * nw[t] * w0[t];
    for (int d = 16; d; d >>= 1) c += __shfl_xor(c, d, 32);
    if ((t & 31) == 0) red[t >> 5] = c;
    __syncthreads();
    if (t == 0) {
        float s = 0.0f;
        for (int i = 0; i < 8; ++i) s += red[i];
        s += b0[0];
        out[row] = (s >= 0.0f) ? s : 0.01f * s;
    }
}

// Deterministic placeholder out_proj weights, written directly as bf16.
__global__ void init_outw_kernel(bf16_t* __restrict__ w, int n)
{
    int i = blockIdx.x * blockDim.x + threadIdx.x;
    if (i < n) {
        unsigned s = (unsigned)i * 2654435761u ^ 0x9E3779B9u;
        s ^= s >> 16; s *= 0x85EBCA6Bu; s ^= s >> 13;
        float u = (float)(s & 0xFFFFFFu) * (1.0f / 16777216.0f) - 0.5f;
        w[i] = (bf16_t)(u * 0.069f);
    }
}

// ---------------------------------------------------------------------------
extern "C" void kernel_launch(void* const* d_in, const int* in_sizes, int n_in,
                              void* d_out, int out_size, void* d_ws, size_t ws_size,
                              hipStream_t stream)
{
    (void)in_sizes; (void)n_in; (void)out_size; (void)ws_size;

    const float* x        = (const float*)d_in[0];
    const float* lin_in_w = (const float*)d_in[1];
    const float* lin_in_b = (const float*)d_in[2];
    const float* in_proj  = (const float*)d_in[3];
    const float* conv_w   = (const float*)d_in[4];
    const float* conv_b   = (const float*)d_in[5];
    const float* xp_w     = (const float*)d_in[6];
    const float* dt_w     = (const float*)d_in[7];
    const float* dt_b     = (const float*)d_in[8];
    const float* A_log    = (const float*)d_in[9];
    const float* Dp       = (const float*)d_in[10];
    const float* norm_w   = (const float*)d_in[11];
    const float* norm_f   = (const float*)d_in[12];
    const float* lo_w     = (const float*)d_in[13];
    const float* lo_b     = (const float*)d_in[14];
    float* out = (float*)d_out;

    // ---- workspace: fp32 region then bf16 region (all 16B aligned) ----
    float* wsf  = (float*)d_ws;
    float* h    = wsf;  wsf += (size_t)ROWS * DM;       // fp32
    float* xz   = wsf;  wsf += (size_t)ROWS * 2 * DI;
    float* xsf  = wsf;  wsf += (size_t)ROWS * DI;
    float* xdbl = wsf;  wsf += (size_t)ROWS * 48;
    float* dlt  = wsf;  wsf += (size_t)ROWS * DI;

    bf16_t* wsb   = (bf16_t*)wsf;
    bf16_t* x_bf  = wsb;  wsb += (size_t)ROWS * 32;     // activations (bf16)
    bf16_t* xn_bf = wsb;  wsb += (size_t)ROWS * DM;
    bf16_t* xs_bf = wsb;  wsb += (size_t)ROWS * DI;
    bf16_t* xd_bf = wsb;  wsb += (size_t)ROWS * 48;
    bf16_t* yb_bf = wsb;  wsb += (size_t)ROWS * DI;
    bf16_t* w_lin = wsb;  wsb += (size_t)DM * 32;       // weights (bf16, padded)
    bf16_t* w_inp = wsb;  wsb += (size_t)4 * 2 * DI * DM;
    bf16_t* w_xp  = wsb;  wsb += (size_t)4 * 64 * DI;   // 48 -> 64 rows
    bf16_t* w_dt  = wsb;  wsb += (size_t)4 * DI * 32;   // K 16 -> 32
    bf16_t* w_out = wsb;  wsb += (size_t)4 * DM * DI;

    auto cvt = [&](const float* s, bf16_t* d, int N, int K, int Np, int Kp) {
        int n = Np * Kp;
        cvt_weight_kernel<<<(n + 255) / 256, 256, 0, stream>>>(s, d, N, K, Np, Kp);
    };

    // ---- one-time pre-pass: bf16 weight (and input) conversion ----
    cvt(x, x_bf, ROWS, 32, ROWS, 32);
    cvt(lin_in_w, w_lin, DM, 32, DM, 32);
    cvt(in_proj, w_inp, 4 * 2 * DI, DM, 4 * 2 * DI, DM);   // layers contiguous
    for (int i = 0; i < 4; ++i) {
        cvt(xp_w + (size_t)i * 48 * DI, w_xp + (size_t)i * 64 * DI, 48, DI, 64, DI);
        cvt(dt_w + (size_t)i * DI * 16, w_dt + (size_t)i * DI * 32, DI, 16, DI, 32);
    }
    { int n = 4 * DM * DI;
      init_outw_kernel<<<(n + 255) / 256, 256, 0, stream>>>(w_out, n); }

    auto gemm = [&](const bf16_t* A, int lda, const bf16_t* B, int ldb,
                    const float* bias, const float* add, float* C, int ldc,
                    bf16_t* Cbf, int M, int N, int K, int act) {
        dim3 grid((N + 63) / 64, (M + 127) / 128);
        gemm_bf16_wmma<<<grid, 256, 0, stream>>>(A, lda, B, ldb, bias, add,
                                                 C, ldc, Cbf, M, N, K, act);
    };

    // h = x @ lin_in_w.T + lin_in_b            (8192 x 256, K=32)
    gemm(x_bf, 32, w_lin, 32, lin_in_b, nullptr, h, DM, nullptr, ROWS, DM, 32, 0);

    for (int i = 0; i < 4; ++i) {
        // xn = rmsnorm(h) * norm_w[i]  (bf16)
        rmsnorm_kernel<<<ROWS, DM, 0, stream>>>(h, norm_w + (size_t)i * DM, xn_bf);
        // xz = xn @ in_proj[i].T               (8192 x 1024, K=256)
        gemm(xn_bf, DM, w_inp + (size_t)i * 2 * DI * DM, DM, nullptr, nullptr,
             xz, 2 * DI, nullptr, ROWS, 2 * DI, DM, 0);
        // xs = silu(causal_conv(xz[:, :DI]) + conv_b)  (fp32 + bf16)
        conv_silu_kernel<<<(ROWS * DI) / 256, 256, 0, stream>>>(
            xz, conv_w + (size_t)i * DI * 4, conv_b + (size_t)i * DI, xsf, xs_bf);
        // xdbl = xs @ xp_w[i].T                (8192 x 48, K=512; fp32 + bf16)
        gemm(xs_bf, DI, w_xp + (size_t)i * 64 * DI, DI, nullptr, nullptr,
             xdbl, 48, xd_bf, ROWS, 48, DI, 0);
        // delta = softplus(xdbl[:, :16] @ dt_w[i].T + dt_b[i])
        // (K padded to 32: A cols 16..31 hit zero weight columns)
        gemm(xd_bf, 48, w_dt + (size_t)i * DI * 32, 32, dt_b + (size_t)i * DI, nullptr,
             dlt, DI, nullptr, ROWS, DI, 32, 1);
        // yb = gated selective scan (bf16)
        scan_kernel<<<16, 256, 0, stream>>>(
            xsf, dlt, xdbl, xz, A_log + (size_t)i * DI * DSTATE, Dp + (size_t)i * DI, yb_bf);
        // h = h + yb @ outw[i].T               (8192 x 256, K=512, residual fused)
        gemm(yb_bf, DI, w_out + (size_t)i * DM * DI, DI, nullptr, h,
             h, DM, nullptr, ROWS, DM, DI, 0);
    }

    // out = leaky_relu(rmsnorm(h) * norm_f @ lo_w.T + lo_b)
    final_kernel<<<ROWS, DM, 0, stream>>>(h, norm_f, lo_w, lo_b, out);
}
